// DiffusionPropers_82841329205439
// MI455X (gfx1250) — matrix operand order repaired and checked by reading, not compiled, not run on hardware
//
#include <hip/hip_runtime.h>
#include <math.h>

// ---------------------------------------------------------------------------
// DiffusionPropers on MI455X (gfx1250)
//   Key identity: the T=4 rows of a proper share 512/516 features, so the
//   dominant K=512 part of layer 0 is computed once per PROPER:
//     h_pre[p]   = enc_cat[p] @ W0[0:512]            (16 props per block)
//     h0[p,t]    = leaky(h_pre[p] + extras[p,t] @ W0[512:516] + b0)
//   FLOPs: ~79 -> ~41 GFLOP; weight L2 traffic halves (64 rows share one
//   207KB B-panel stream).  All GEMMs are v_wmma_f32_16x16x32_bf16 with f32
//   accumulate; weights/encoded pre-converted to bf16 (L2-resident).
//   Block = 1 wave (wave32), 16 propers = 64 (p,t) rows.
//   Round 5: sin/cos computed algebraically from the atan2 operands
//   (no libm transcendentals), LDS overlays tightened to 44KB so 7 blocks
//   fit per 320KB WGP.
// ---------------------------------------------------------------------------

#define LEAKY 0.001f
#define DIN   516
#define DINP  544           // padded K for layer 0 (extras at 512..515)
#define DH    128

typedef __attribute__((ext_vector_type(2)))  unsigned int u32x2;
typedef __attribute__((ext_vector_type(4)))  unsigned int u32x4;
typedef __attribute__((ext_vector_type(8)))  unsigned int u32x8;
typedef __attribute__((ext_vector_type(16))) __bf16      v16bf;
typedef __attribute__((ext_vector_type(8)))  float       v8f;

static __device__ __forceinline__ unsigned short f2bf(float f) {
  unsigned int u = __builtin_bit_cast(unsigned int, f);
  u += 0x7FFFu + ((u >> 16) & 1u);           // round-to-nearest-even
  return (unsigned short)(u >> 16);
}
static __device__ __forceinline__ v16bf mkfrag(u32x4 lo, u32x4 hi) {
  u32x8 u;
  u[0]=lo[0]; u[1]=lo[1]; u[2]=lo[2]; u[3]=lo[3];
  u[4]=hi[0]; u[5]=hi[1]; u[6]=hi[2]; u[7]=hi[3];
  return __builtin_bit_cast(v16bf, u);
}
// A fragment (16x32 bf16, row-major source, rows 16B aligned).
// Lanes 0-15 row m hold K k0..k0+15; lanes 16-31 hold K k0+16..k0+31.
static __device__ __forceinline__ v16bf load_a_frag(const unsigned short* base,
                                                    int strideElems, int k0, int lane) {
  const int m = lane & 15, sel = lane >> 4;
  const unsigned short* r = base + m * strideElems + k0 + 16 * sel;
  u32x4 lo = *(const u32x4*)(r);
  u32x4 hi = *(const u32x4*)(r + 8);
  return mkfrag(lo, hi);
}
// B fragment (32x16 bf16) from N-major weight Wt[n][k]:
// lane (n = col0 + lane&15) holds K = k0+16*sel .. +15, contiguous 32B.
static __device__ __forceinline__ v16bf load_b_frag(const unsigned short* Wt,
                                                    int kstride, int col0, int k0, int lane) {
  const int n    = col0 + (lane & 15);
  const int koff = k0 + 16 * (lane >> 4);
  const u32x4* p = (const u32x4*)(Wt + n * kstride + koff);
  return mkfrag(p[0], p[1]);
}
static __device__ __forceinline__ v8f wmma_bf16(v16bf a, v16bf b, v8f c) {
  return __builtin_amdgcn_wmma_f32_16x16x32_bf16(false, a, false, b, (short)0, c,
                                                 false, false);
}
// C/D layout: VGPR r -> row r (lanes 0-15) / r+8 (lanes 16-31), lane -> column.
static __device__ __forceinline__ void store_tile(v8f acc, unsigned short* dst,
                                                  int col0, float bias, int lane) {
  const int n   = col0 + (lane & 15);
  const int sel = lane >> 4;
#pragma unroll
  for (int r = 0; r < 8; ++r) {
    float v = acc[r] + bias;
    v = v > 0.f ? v : v * LEAKY;
    dst[(r + 8 * sel) * DH + n] = f2bf(v);
  }
}
static __device__ __forceinline__ void store_pre(v8f acc, float* dst, int col0, int lane) {
  const int n = col0 + (lane & 15), sel = lane >> 4;
#pragma unroll
  for (int r = 0; r < 8; ++r) dst[(r + 8 * sel) * DH + n] = acc[r];
}
// h0[row][n] = leaky(accExtras + h_pre[row>>2][n] + b0[n])
static __device__ __forceinline__ void store_h0(v8f acc, const float* pre,
                                                unsigned short* dst, int rowbase,
                                                int col0, float bias, int lane) {
  const int n = col0 + (lane & 15), sel = lane >> 4;
#pragma unroll
  for (int r = 0; r < 8; ++r) {
    const int row = rowbase + r + 8 * sel;
    float v = acc[r] + pre[(row >> 2) * DH + n] + bias;
    v = v > 0.f ? v : v * LEAKY;
    dst[row * DH + n] = f2bf(v);
  }
}

// 128->128 layer over 64 rows: col-quarter passes (2 col tiles x 4 row tiles
// = 64 acc VGPRs live); every B fragment loaded exactly once, feeds 4 WMMAs.
static __device__ __forceinline__ void layer64(const unsigned short* src,
                                               const unsigned short* Wt,
                                               const float* bias, unsigned short* dst,
                                               int lane) {
#pragma unroll 1
  for (int cq = 0; cq < 4; ++cq) {
    v8f acc[4][2];
#pragma unroll
    for (int rt = 0; rt < 4; ++rt)
#pragma unroll
      for (int c = 0; c < 2; ++c) acc[rt][c] = (v8f){0,0,0,0,0,0,0,0};
#pragma unroll 1
    for (int kc = 0; kc < 4; ++kc) {
      const int k0 = kc * 32;
      v16bf a0 = load_a_frag(src,           DH, k0, lane);
      v16bf a1 = load_a_frag(src + 16 * DH, DH, k0, lane);
      v16bf a2 = load_a_frag(src + 32 * DH, DH, k0, lane);
      v16bf a3 = load_a_frag(src + 48 * DH, DH, k0, lane);
#pragma unroll
      for (int c = 0; c < 2; ++c) {
        v16bf b = load_b_frag(Wt, DH, cq * 32 + c * 16, k0, lane);
        acc[0][c] = wmma_bf16(a0, b, acc[0][c]);
        acc[1][c] = wmma_bf16(a1, b, acc[1][c]);
        acc[2][c] = wmma_bf16(a2, b, acc[2][c]);
        acc[3][c] = wmma_bf16(a3, b, acc[3][c]);
      }
    }
#pragma unroll
    for (int c = 0; c < 2; ++c) {
      const float bi = bias[cq * 32 + c * 16 + (lane & 15)];
#pragma unroll
      for (int rt = 0; rt < 4; ++rt)
        store_tile(acc[rt][c], dst + rt * 16 * DH, cq * 32 + c * 16, bi, lane);
    }
  }
}

// Transpose + bf16-convert weights: W*t[n][k] = bf16(W*[k][n]); zero-pad
// k>=516 (W0t) and n>=2 (W3t).
__global__ void prep_weights(const float* __restrict__ W0, const float* __restrict__ W1,
                             const float* __restrict__ W2, const float* __restrict__ W3,
                             unsigned short* __restrict__ W0t,
                             unsigned short* __restrict__ W1t,
                             unsigned short* __restrict__ W2t,
                             unsigned short* __restrict__ W3t) {
  const int i = blockIdx.x * blockDim.x + threadIdx.x;
  if (i < 128 * DINP) {
    const int n = i / DINP, k = i % DINP;
    W0t[i] = (k < DIN) ? f2bf(W0[k * 128 + n]) : (unsigned short)0;
  }
  if (i < 128 * 128) {
    const int n = i / 128, k = i % 128;
    W1t[i] = f2bf(W1[k * 128 + n]);
    W2t[i] = f2bf(W2[k * 128 + n]);
  }
  if (i < 16 * 128) {
    const int n = i / 128, k = i % 128;
    W3t[i] = (n < 2) ? f2bf(W3[k * 2 + n]) : (unsigned short)0;
  }
}

// Pre-convert the gather table to bf16 (6.4MB; L2-resident afterwards).
__global__ void prep_encoded(const float* __restrict__ enc,
                             unsigned short* __restrict__ encB, int n) {
  const int i = blockIdx.x * blockDim.x + threadIdx.x;
  if (i < n) encB[i] = f2bf(enc[i]);
}

__global__ __launch_bounds__(32)
void mlp_kernel(const float* __restrict__ coords, const long long* __restrict__ propers,
                const unsigned short* __restrict__ encB, const float* __restrict__ tvals,
                const unsigned short* __restrict__ W0t,
                const unsigned short* __restrict__ W1t,
                const unsigned short* __restrict__ W2t,
                const unsigned short* __restrict__ W3t,
                const float* __restrict__ b0, const float* __restrict__ b1,
                const float* __restrict__ b2, const float* __restrict__ b3,
                float* __restrict__ out, int P) {
  // LDS carve (44KB total; 7 blocks fit per 320KB WGP):
  //   sEnc dead after layer0-pre  -> sH1 overlays it,
  //   sEx  dead after extras GEMM -> sD  overlays it.
  __shared__ __align__(16) unsigned char smem[45056];
  unsigned short* sEnc = (unsigned short*)(smem);           // [16][512] 16KB
  unsigned short* sH1  = (unsigned short*)(smem);           // [64][128] 16KB (overlay)
  unsigned short* sEx  = (unsigned short*)(smem + 16384);   // [64][32]   4KB
  float*          sD   = (float*)         (smem + 16384);   // [64][16]   4KB (overlay)
  float*          sPre = (float*)         (smem + 20480);   // [16][128]  8KB
  unsigned short* sH0  = (unsigned short*)(smem + 28672);   // [64][128] 16KB

  const int lane      = threadIdx.x;
  const int totalRows = P * 4;
  const int row0      = blockIdx.x * 64;
  const int pbase     = blockIdx.x * 16;      // 16 propers per block

  // -------- geometry extras: 2 (p,t) rows per lane -> sEx[64][32] --------
#pragma unroll 1
  for (int jr = 0; jr < 2; ++jr) {
    const int rl = lane + 32 * jr;
    int rowg = row0 + rl; if (rowg >= totalRows) rowg = totalRows - 1;
    const int pg = rowg >> 2, tix = rowg & 3;
    const long long* pr = propers + (long long)pg * 4;
    const int i0 = (int)pr[0], i1 = (int)pr[1], i2 = (int)pr[2], i3 = (int)pr[3];
    const int o0 = i0 * 12 + tix * 3, o1 = i1 * 12 + tix * 3;
    const int o2 = i2 * 12 + tix * 3, o3 = i3 * 12 + tix * 3;
    const float c0x = coords[o0], c0y = coords[o0 + 1], c0z = coords[o0 + 2];
    const float c1x = coords[o1], c1y = coords[o1 + 1], c1z = coords[o1 + 2];
    const float c2x = coords[o2], c2y = coords[o2 + 1], c2z = coords[o2 + 2];
    const float c3x = coords[o3], c3y = coords[o3 + 1], c3z = coords[o3 + 2];
    const float u1x = c1x - c0x, u1y = c1y - c0y, u1z = c1z - c0z;
    const float u2x = c2x - c1x, u2y = c2y - c1y, u2z = c2z - c1z;
    const float u3x = c3x - c2x, u3y = c3y - c2y, u3z = c3z - c2z;
    const float ax = u1y * u2z - u1z * u2y;     // u1 x u2
    const float ay = u1z * u2x - u1x * u2z;
    const float az = u1x * u2y - u1y * u2x;
    const float bx = u2y * u3z - u2z * u3y;     // u2 x u3
    const float by = u2z * u3x - u2x * u3z;
    const float bz = u2x * u3y - u2y * u3x;
    const float u2n = sqrtf(u2x * u2x + u2y * u2y + u2z * u2z);
    const float yv = u2n * (u1x * bx + u1y * by + u1z * bz);
    const float xv = ax * bx + ay * by + az * bz;
    // sin/cos of atan2(yv, xv) without transcendentals:
    //   sin = y * rsqrt(x^2+y^2), cos = x * rsqrt(x^2+y^2);
    //   degenerate (x=y=0) matches atan2(0,0)=0 -> sin=0, cos=1.
    const float r2   = xv * xv + yv * yv;
    const float rinv = rsqrtf(fmaxf(r2, 1e-30f));
    const float st   = (r2 > 0.f) ? yv * rinv : 0.f;
    const float ct   = (r2 > 0.f) ? xv * rinv : 1.f;
    const float drx = c0x - c3x, dry = c0y - c3y, drz = c0z - c3z;
    const float dl  = sqrtf(fmaxf(drx * drx + dry * dry + drz * drz, 1e-12f));

    unsigned short* ex = sEx + rl * 32;        // K panel: t,sin,cos,dl,0...
    ex[0] = f2bf(tvals[tix]);
    ex[1] = f2bf(st);
    ex[2] = f2bf(ct);
    ex[3] = f2bf(dl);
#pragma unroll
    for (int j = 4; j < 32; ++j) ex[j] = 0;
  }

  // -------- gather bf16 encoded rows (no T replication): sEnc[16][512] ----
  const int eoff = lane * 4;                   // 4 bf16 = 8B per lane
#pragma unroll 1
  for (int q = 0; q < 16; ++q) {
    int pq = pbase + q; if (pq >= P) pq = P - 1;
    const long long* prq = propers + (long long)pq * 4;
#pragma unroll
    for (int a = 0; a < 4; ++a) {
      const u32x2 v = *(const u32x2*)(encB + (long long)((int)prq[a]) * 128 + eoff);
      *(u32x2*)(sEnc + q * 512 + a * 128 + eoff) = v;
    }
  }
  __syncthreads();

  // -------- layer-0 pre: 16 props x K=512 -> sPre[16][128] (f32, no bias) --
  {
    v8f acc[8];
#pragma unroll
    for (int c = 0; c < 8; ++c) acc[c] = (v8f){0,0,0,0,0,0,0,0};
#pragma unroll 1
    for (int kc = 0; kc < 16; ++kc) {
      const int k0 = kc * 32;
      v16bf a = load_a_frag(sEnc, 512, k0, lane);
#pragma unroll
      for (int c = 0; c < 8; ++c) {
        v16bf b = load_b_frag(W0t, DINP, c * 16, k0, lane);
        acc[c] = wmma_bf16(a, b, acc[c]);
      }
    }
#pragma unroll
    for (int c = 0; c < 8; ++c) store_pre(acc[c], sPre, c * 16, lane);
  }
  __syncthreads();

  // -------- extras GEMM (K=32, W0t cols 512..543) + combine -> sH0 --------
#pragma unroll 1
  for (int rt = 0; rt < 4; ++rt) {
    v8f acc[8];
#pragma unroll
    for (int c = 0; c < 8; ++c) acc[c] = (v8f){0,0,0,0,0,0,0,0};
    v16bf a = load_a_frag(sEx + rt * 16 * 32, 32, 0, lane);
#pragma unroll
    for (int c = 0; c < 8; ++c) {
      v16bf b = load_b_frag(W0t, DINP, c * 16, 512, lane);
      acc[c] = wmma_bf16(a, b, acc[c]);
    }
#pragma unroll
    for (int c = 0; c < 8; ++c) {
      const float bi = b0[c * 16 + (lane & 15)];
      store_h0(acc[c], sPre, sH0, rt * 16, c * 16, bi, lane);
    }
  }
  __syncthreads();

  // -------- layers 1 & 2 over 64 rows (sH1 overlays dead sEnc) --------
  layer64(sH0, W1t, b1, sH1, lane);
  __syncthreads();
  layer64(sH1, W2t, b2, sH0, lane);
  __syncthreads();

  // -------- layer 3 (128 -> 2, padded to 16 cols) --------
  {
    v8f acc[4];
#pragma unroll
    for (int rt = 0; rt < 4; ++rt) acc[rt] = (v8f){0,0,0,0,0,0,0,0};
#pragma unroll 1
    for (int kc = 0; kc < 4; ++kc) {
      const int k0 = kc * 32;
      v16bf b = load_b_frag(W3t, DH, 0, k0, lane);
#pragma unroll
      for (int rt = 0; rt < 4; ++rt) {
        v16bf a = load_a_frag(sH0 + rt * 16 * DH, DH, k0, lane);
        acc[rt] = wmma_bf16(a, b, acc[rt]);
      }
    }
    const int n = lane & 15, sel = lane >> 4;
#pragma unroll
    for (int rt = 0; rt < 4; ++rt)
#pragma unroll
      for (int r = 0; r < 8; ++r)
        sD[(rt * 16 + r + 8 * sel) * 16 + n] = acc[rt][r];
  }
  __syncthreads();

  // -------- scatter: 2 rows per lane, atomics onto copy of `answer` --------
#pragma unroll 1
  for (int jr = 0; jr < 2; ++jr) {
    const int rl = lane + 32 * jr;
    const int rowg = row0 + rl;
    if (rowg < totalRows) {
      const float d0 = sD[rl * 16 + 0] + b3[0];
      const float d1 = sD[rl * 16 + 1] + b3[1];
      const int pg = rowg >> 2, tix = rowg & 3;
      const long long* pr = propers + (long long)pg * 4;
      const int i0 = (int)pr[0], i3 = (int)pr[3];
      const int o0 = i0 * 12 + tix * 3, o3 = i3 * 12 + tix * 3;
      const float drx = coords[o0]     - coords[o3];
      const float dry = coords[o0 + 1] - coords[o3 + 1];
      const float drz = coords[o0 + 2] - coords[o3 + 2];
      const float dl  = sqrtf(fmaxf(drx * drx + dry * dry + drz * drz, 1e-12f));
      const float inv = 1.0f / dl;
      const float s0 = -0.5f * d0, s1 = 0.5f * d1;
      const int ob0 = (i0 * 4 + tix) * 3;
      atomicAdd(&out[ob0 + 0], s0 * drx * inv);
      atomicAdd(&out[ob0 + 1], s0 * dry * inv);
      atomicAdd(&out[ob0 + 2], s0 * drz * inv);
      const int ob3 = (i3 * 4 + tix) * 3;
      atomicAdd(&out[ob3 + 0], s1 * drx * inv);
      atomicAdd(&out[ob3 + 1], s1 * dry * inv);
      atomicAdd(&out[ob3 + 2], s1 * drz * inv);
    }
  }
}

extern "C" void kernel_launch(void* const* d_in, const int* in_sizes, int n_in,
                              void* d_out, int out_size, void* d_ws, size_t ws_size,
                              hipStream_t stream) {
  const float*     coords  = (const float*)d_in[0];
  const long long* propers = (const long long*)d_in[1];
  const float*     encoded = (const float*)d_in[2];
  const float*     tvals   = (const float*)d_in[3];
  const float*     answer  = (const float*)d_in[4];
  const float*     W0      = (const float*)d_in[5];
  const float*     b0      = (const float*)d_in[6];
  const float*     W1      = (const float*)d_in[7];
  const float*     b1      = (const float*)d_in[8];
  const float*     W2      = (const float*)d_in[9];
  const float*     b2      = (const float*)d_in[10];
  const float*     W3      = (const float*)d_in[11];
  const float*     b3      = (const float*)d_in[12];

  const int P  = in_sizes[1] / 4;
  const int NE = in_sizes[2];                 // N * 128 encoded elements

  unsigned short* W0t  = (unsigned short*)d_ws;        // 128*544 bf16
  unsigned short* W1t  = W0t + 128 * DINP;             // 128*128 bf16
  unsigned short* W2t  = W1t + 128 * 128;              // 128*128 bf16
  unsigned short* W3t  = W2t + 128 * 128;              // 16*128  bf16
  unsigned short* encB = W3t + 16 * 128;               // NE      bf16

  // out starts as a copy of `answer` (scatter-adds land on top of it)
  hipMemcpyAsync(d_out, answer, (size_t)out_size * sizeof(float),
                 hipMemcpyDeviceToDevice, stream);

  prep_weights<<<(128 * DINP + 255) / 256, 256, 0, stream>>>(W0, W1, W2, W3,
                                                             W0t, W1t, W2t, W3t);
  prep_encoded<<<(NE + 255) / 256, 256, 0, stream>>>(encoded, encB, NE);

  const int rows = P * 4;
  const int nblk = (rows + 63) / 64;
  mlp_kernel<<<nblk, 32, 0, stream>>>(coords, propers, encB, tvals,
                                      W0t, W1t, W2t, W3t, b0, b1, b2, b3,
                                      (float*)d_out, P);
}
